// Attention_88012469829981
// MI455X (gfx1250) — compile-verified, matrix operand order checked by reading, not run on hardware
//
#include <hip/hip_runtime.h>

// ---------------------------------------------------------------------------
// GPT-2 attention block for MI455X (gfx1250, wave32, WMMA f32<-f16)
// B=2, S=2048, D=1024, H=16, hd=64
// ---------------------------------------------------------------------------

#define BATCH 2
#define SEQ   2048
#define DMODEL 1024
#define NHEAD 16
#define HDIM  64
#define SCALE 0.125f           // 1/sqrt(64)
#define MASKED_BIAS -10000.0f

typedef __attribute__((ext_vector_type(4)))  _Float16 h4;
typedef __attribute__((ext_vector_type(8)))  _Float16 h8;
typedef __attribute__((ext_vector_type(16))) _Float16 h16;
typedef __attribute__((ext_vector_type(8)))  float    f8;

__device__ __forceinline__ f8 fzero() {
  f8 z;
#pragma unroll
  for (int i = 0; i < 8; ++i) z[i] = 0.0f;
  return z;
}

// WMMA f32 += f16 A(16x32) * f16 B(32x16)
__device__ __forceinline__ f8 wmma_f16(h16 a, h16 b, f8 c) {
  return __builtin_amdgcn_wmma_f32_16x16x32_f16(
      /*neg_a=*/false, a, /*neg_b=*/false, b,
      /*c_mod=*/(short)0, c, /*reuse_a=*/false, /*reuse_b=*/false);
}

// Load a 16x32 f16 fragment (A-layout; also used for B when the source is
// stored "outer = N, inner = K", i.e. transposed). CDNA5 16-bit A layout:
//   lane holds row (lane & 15); element e covers K = kb + 16*(e/8) + e%8,
//   kb = 8*(lane>>4).  => two contiguous 16B chunks per lane.
__device__ __forceinline__ h16 frag_ld(const _Float16* p, int ld) {
  const int lane = threadIdx.x & 31;
  const _Float16* q = p + (lane & 15) * ld + ((lane >> 4) << 3);
  h8 lo = *(const h8*)(q);
  h8 hi = *(const h8*)(q + 16);
  h16 out;
#pragma unroll
  for (int i = 0; i < 8; ++i) { out[i] = lo[i]; out[i + 8] = hi[i]; }
  return out;
}

// ---------------------------------------------------------------------------
// Kernel 1: QKV GEMM.  X[4096,1024] (f32) @ W[1024,3072] (f32) + b
// Tile 128(M) x 128(N) x 32(K), 8 waves; wave -> 32x64 (8 WMMA / K-step).
// Next K-tile staged in registers while WMMAs run (double buffering).
// Scatters f16 results into Qh[b,h,s,d], Kh[b,h,s,d], Vt[b,h,d,s].
// ---------------------------------------------------------------------------
__global__ __launch_bounds__(256) void qkv_gemm_kernel(
    const float* __restrict__ X, const float* __restrict__ W,
    const float* __restrict__ bias,
    _Float16* __restrict__ Qh, _Float16* __restrict__ Kh,
    _Float16* __restrict__ Vt)
{
  __shared__ __align__(16) _Float16 As [128 * 32];   // [m][k]
  __shared__ __align__(16) _Float16 Bst[128 * 32];   // [n][k] (transposed)

  const int tid  = threadIdx.x;
  const int lane = tid & 31;
  const int wave = tid >> 5;
  const int m0 = blockIdx.y * 128;
  const int n0 = blockIdx.x * 128;
  const int wm = wave & 3;           // 0..3 -> rows wm*32
  const int wn = wave >> 2;          // 0..1 -> cols wn*64

  f8 acc[2][4];
#pragma unroll
  for (int i = 0; i < 2; ++i)
#pragma unroll
    for (int j = 0; j < 4; ++j) acc[i][j] = fzero();

  float4 ra[4], rb[4];
  // --- prologue: stage K-tile 0 into registers ---
#pragma unroll
  for (int it = 0; it < 4; ++it) {
    const int i = tid * 4 + it * 1024;
    ra[it] = *(const float4*)(X + (size_t)(m0 + (i >> 5)) * DMODEL + (i & 31));
  }
#pragma unroll
  for (int it = 0; it < 4; ++it) {
    const int i = tid * 4 + it * 1024;
    rb[it] = *(const float4*)(W + (size_t)(i >> 7) * 3072 + n0 + (i & 127));
  }

  for (int kk = 0; kk < DMODEL; kk += 32) {
    // --- commit staged registers to LDS (f32 -> f16; B transposed) ---
#pragma unroll
    for (int it = 0; it < 4; ++it) {
      const int i = tid * 4 + it * 1024;
      const int row = i >> 5, col = i & 31;
      As[row * 32 + col + 0] = (_Float16)ra[it].x;
      As[row * 32 + col + 1] = (_Float16)ra[it].y;
      As[row * 32 + col + 2] = (_Float16)ra[it].z;
      As[row * 32 + col + 3] = (_Float16)ra[it].w;
    }
#pragma unroll
    for (int it = 0; it < 4; ++it) {
      const int i = tid * 4 + it * 1024;
      const int row = i >> 7, col = i & 127;   // row=k, col=n
      Bst[(col + 0) * 32 + row] = (_Float16)rb[it].x;
      Bst[(col + 1) * 32 + row] = (_Float16)rb[it].y;
      Bst[(col + 2) * 32 + row] = (_Float16)rb[it].z;
      Bst[(col + 3) * 32 + row] = (_Float16)rb[it].w;
    }
    __syncthreads();

    // --- stage next K-tile (loads fly during the WMMAs below) ---
    if (kk + 32 < DMODEL) {
#pragma unroll
      for (int it = 0; it < 4; ++it) {
        const int i = tid * 4 + it * 1024;
        ra[it] = *(const float4*)(X + (size_t)(m0 + (i >> 5)) * DMODEL +
                                  kk + 32 + (i & 31));
      }
#pragma unroll
      for (int it = 0; it < 4; ++it) {
        const int i = tid * 4 + it * 1024;
        rb[it] = *(const float4*)(W + (size_t)(kk + 32 + (i >> 7)) * 3072 +
                                  n0 + (i & 127));
      }
    }

    // --- 8 WMMAs from LDS fragments ---
    const h16 a0 = frag_ld(&As[(wm * 32 +  0) * 32], 32);
    const h16 a1 = frag_ld(&As[(wm * 32 + 16) * 32], 32);
#pragma unroll
    for (int fn = 0; fn < 4; ++fn) {
      const h16 b = frag_ld(&Bst[(wn * 64 + fn * 16) * 32], 32);
      acc[0][fn] = wmma_f16(a0, b, acc[0][fn]);
      acc[1][fn] = wmma_f16(a1, b, acc[1][fn]);
    }
    __syncthreads();
  }

  // --- epilogue: bias + f16 scatter (branch is uniform per block) ---
  const int colL = lane & 15;
  const int rowL = (lane >> 4) << 3;
#pragma unroll
  for (int fm = 0; fm < 2; ++fm)
#pragma unroll
    for (int fn = 0; fn < 4; ++fn)
#pragma unroll
      for (int r = 0; r < 8; ++r) {
        const int grow = m0 + wm * 32 + fm * 16 + r + rowL;  // token index
        const int gcol = n0 + wn * 64 + fn * 16 + colL;      // qkv column
        const float val = acc[fm][fn][r] + bias[gcol];
        const _Float16 hv = (_Float16)val;
        const int b = grow >> 11, s = grow & 2047;
        const int which = gcol >> 10, nn = gcol & 1023;
        const int h = nn >> 6, d = nn & 63;
        if (which == 0)
          Qh[(((size_t)(b * NHEAD + h) * SEQ) + s) * HDIM + d] = hv;
        else if (which == 1)
          Kh[(((size_t)(b * NHEAD + h) * SEQ) + s) * HDIM + d] = hv;
        else
          Vt[(((size_t)(b * NHEAD + h) * HDIM) + d) * SEQ + s] = hv;
      }
}

// ---------------------------------------------------------------------------
// Kernel 2: flash attention, one wave per (b, h, 16-query tile).
// Per 32-key block: K and V fragments loaded up-front (V loads overlap the
// softmax VALU chain), 4 WMMA (QK^T) + online softmax + 4 WMMA (PV).
// P converted D-layout -> A-layout through a private LDS slice.
// ---------------------------------------------------------------------------
__global__ __launch_bounds__(256) void flash_attn_kernel(
    const _Float16* __restrict__ Qh, const _Float16* __restrict__ Kh,
    const _Float16* __restrict__ Vt, _Float16* __restrict__ Of)
{
  __shared__ __align__(16) _Float16 Plds[8 * 16 * 32];   // 1 KB per wave

  const int lane = threadIdx.x & 31;
  const int wave = threadIdx.x >> 5;
  const int gid = blockIdx.x * 8 + wave;       // 0..4095
  const int qtile = gid & 127;
  const int h = (gid >> 7) & 15;
  const int b = gid >> 11;
  const int q0 = qtile * 16;

  const int colk = lane & 15;
  const int rowL = (lane >> 4) << 3;

  const _Float16* Qp = Qh + ((size_t)(b * NHEAD + h) * SEQ + q0) * HDIM;
  const h16 aq0 = frag_ld(Qp, HDIM);           // hd dims  0..31
  const h16 aq1 = frag_ld(Qp + 32, HDIM);      // hd dims 32..63

  float m8[8], l8[8];
#pragma unroll
  for (int r = 0; r < 8; ++r) { m8[r] = -3.0e38f; l8[r] = 0.0f; }
  f8 o4[4];
#pragma unroll
  for (int n = 0; n < 4; ++n) o4[n] = fzero();

  _Float16* pw = &Plds[wave * (16 * 32)];
  const _Float16* Vp = Vt + (size_t)(b * NHEAD + h) * HDIM * SEQ;

  const int kend = q0 + 16;                    // causal limit (exclusive)
  for (int kb = 0; kb < kend; kb += 32) {
    const _Float16* Kp = Kh + ((size_t)(b * NHEAD + h) * SEQ + kb) * HDIM;

    // ---- fragment loads up-front: 4x K, 4x V (all independent) ----
    const h16 k00 = frag_ld(Kp, HDIM);
    const h16 k01 = frag_ld(Kp + 32, HDIM);
    const h16 k10 = frag_ld(Kp + 16 * HDIM, HDIM);
    const h16 k11 = frag_ld(Kp + 16 * HDIM + 32, HDIM);
    h16 bv[4];
#pragma unroll
    for (int n = 0; n < 4; ++n)
      bv[n] = frag_ld(Vp + (size_t)(n * 16) * SEQ + kb, SEQ);

    // ---- scores: two 16-key subtiles, contracting over hd=64 ----
    f8 s0 = fzero(), s1 = fzero();
    s0 = wmma_f16(aq0, k00, s0);
    s0 = wmma_f16(aq1, k01, s0);
    s1 = wmma_f16(aq0, k10, s1);
    s1 = wmma_f16(aq1, k11, s1);

    // ---- scale + causal mask + online softmax (rows r, r+8 per lane) ----
#pragma unroll
    for (int r = 0; r < 8; ++r) {
      const int qr = q0 + r + rowL;
      float v0 = s0[r] * SCALE;
      float v1 = s1[r] * SCALE;
      if (kb + colk      > qr) v0 = MASKED_BIAS;
      if (kb + 16 + colk > qr) v1 = MASKED_BIAS;

      float t = fmaxf(v0, v1);                 // row-max across 16 lanes
      t = fmaxf(t, __shfl_xor(t, 1, 32));
      t = fmaxf(t, __shfl_xor(t, 2, 32));
      t = fmaxf(t, __shfl_xor(t, 4, 32));
      t = fmaxf(t, __shfl_xor(t, 8, 32));
      const float m_new = fmaxf(m8[r], t);
      const float alpha = __expf(m8[r] - m_new);
      const float p0 = __expf(v0 - m_new);
      const float p1 = __expf(v1 - m_new);
      float rs = p0 + p1;
      rs += __shfl_xor(rs, 1, 32);
      rs += __shfl_xor(rs, 2, 32);
      rs += __shfl_xor(rs, 4, 32);
      rs += __shfl_xor(rs, 8, 32);
      l8[r] = l8[r] * alpha + rs;
      m8[r] = m_new;
#pragma unroll
      for (int n = 0; n < 4; ++n) o4[n][r] *= alpha;

      // stage P tile (D-layout -> row-major 16x32 in LDS)
      const int prow = r + rowL;
      pw[prow * 32 + colk]      = (_Float16)p0;
      pw[prow * 32 + 16 + colk] = (_Float16)p1;
    }

    // ---- P back as A-fragment, then PV over V^T ----
    const h16 ap = frag_ld(pw, 32);
#pragma unroll
    for (int n = 0; n < 4; ++n) o4[n] = wmma_f16(ap, bv[n], o4[n]);
  }

  // ---- normalize and store merged-head f16 output [b, s, h*64+d] ----
  _Float16* Op = Of + ((size_t)(b * SEQ + q0)) * DMODEL + h * HDIM;
#pragma unroll
  for (int r = 0; r < 8; ++r) {
    const float inv = 1.0f / l8[r];
#pragma unroll
    for (int n = 0; n < 4; ++n) {
      Op[(size_t)(r + rowL) * DMODEL + n * 16 + colk] =
          (_Float16)(o4[n][r] * inv);
    }
  }
}

// ---------------------------------------------------------------------------
// Kernel 3: output projection.  Of16[4096,1024] @ Wp[1024,1024] + bp -> f32
// Same 128x128x32 double-buffered tiling; A is already f16.
// ---------------------------------------------------------------------------
__global__ __launch_bounds__(256) void proj_gemm_kernel(
    const _Float16* __restrict__ A, const float* __restrict__ W,
    const float* __restrict__ bias, float* __restrict__ out)
{
  __shared__ __align__(16) _Float16 As [128 * 32];
  __shared__ __align__(16) _Float16 Bst[128 * 32];

  const int tid  = threadIdx.x;
  const int lane = tid & 31;
  const int wave = tid >> 5;
  const int m0 = blockIdx.y * 128;
  const int n0 = blockIdx.x * 128;
  const int wm = wave & 3;
  const int wn = wave >> 2;

  f8 acc[2][4];
#pragma unroll
  for (int i = 0; i < 2; ++i)
#pragma unroll
    for (int j = 0; j < 4; ++j) acc[i][j] = fzero();

  h8 ra[2];
  float4 rb[4];
#pragma unroll
  for (int it = 0; it < 2; ++it) {
    const int i = tid * 8 + it * 2048;
    ra[it] = *(const h8*)(A + (size_t)(m0 + (i >> 5)) * DMODEL + (i & 31));
  }
#pragma unroll
  for (int it = 0; it < 4; ++it) {
    const int i = tid * 4 + it * 1024;
    rb[it] = *(const float4*)(W + (size_t)(i >> 7) * DMODEL + n0 + (i & 127));
  }

  for (int kk = 0; kk < DMODEL; kk += 32) {
#pragma unroll
    for (int it = 0; it < 2; ++it) {
      const int i = tid * 8 + it * 2048;
      *(h8*)&As[(i >> 5) * 32 + (i & 31)] = ra[it];
    }
#pragma unroll
    for (int it = 0; it < 4; ++it) {
      const int i = tid * 4 + it * 1024;
      const int row = i >> 7, col = i & 127;
      Bst[(col + 0) * 32 + row] = (_Float16)rb[it].x;
      Bst[(col + 1) * 32 + row] = (_Float16)rb[it].y;
      Bst[(col + 2) * 32 + row] = (_Float16)rb[it].z;
      Bst[(col + 3) * 32 + row] = (_Float16)rb[it].w;
    }
    __syncthreads();

    if (kk + 32 < DMODEL) {
#pragma unroll
      for (int it = 0; it < 2; ++it) {
        const int i = tid * 8 + it * 2048;
        ra[it] = *(const h8*)(A + (size_t)(m0 + (i >> 5)) * DMODEL +
                              kk + 32 + (i & 31));
      }
#pragma unroll
      for (int it = 0; it < 4; ++it) {
        const int i = tid * 4 + it * 1024;
        rb[it] = *(const float4*)(W + (size_t)(kk + 32 + (i >> 7)) * DMODEL +
                                  n0 + (i & 127));
      }
    }

    const h16 a0 = frag_ld(&As[(wm * 32 +  0) * 32], 32);
    const h16 a1 = frag_ld(&As[(wm * 32 + 16) * 32], 32);
#pragma unroll
    for (int fn = 0; fn < 4; ++fn) {
      const h16 b = frag_ld(&Bst[(wn * 64 + fn * 16) * 32], 32);
      acc[0][fn] = wmma_f16(a0, b, acc[0][fn]);
      acc[1][fn] = wmma_f16(a1, b, acc[1][fn]);
    }
    __syncthreads();
  }

  const int colL = lane & 15;
  const int rowL = (lane >> 4) << 3;
#pragma unroll
  for (int fm = 0; fm < 2; ++fm)
#pragma unroll
    for (int fn = 0; fn < 4; ++fn)
#pragma unroll
      for (int r = 0; r < 8; ++r) {
        const int grow = m0 + wm * 32 + fm * 16 + r + rowL;
        const int gcol = n0 + wn * 64 + fn * 16 + colL;
        out[(size_t)grow * DMODEL + gcol] = acc[fm][fn][r] + bias[gcol];
      }
}

// ---------------------------------------------------------------------------
extern "C" void kernel_launch(void* const* d_in, const int* in_sizes, int n_in,
                              void* d_out, int out_size, void* d_ws, size_t ws_size,
                              hipStream_t stream) {
  const float* X    = (const float*)d_in[0];   // hidden_states [2,2048,1024]
  const float* Wqkv = (const float*)d_in[1];   // c_attn_w [1024,3072]
  const float* bqkv = (const float*)d_in[2];   // c_attn_b [3072]
  const float* Wp   = (const float*)d_in[3];   // c_proj_w [1024,1024]
  const float* bp   = (const float*)d_in[4];   // c_proj_b [1024]
  float* out = (float*)d_out;

  char* ws = (char*)d_ws;
  _Float16* Qh = (_Float16*)(ws + 0);                    // 8 MB [b,h,s,d]
  _Float16* Kh = (_Float16*)(ws + (size_t)8  * 1048576); // 8 MB [b,h,s,d]
  _Float16* Vt = (_Float16*)(ws + (size_t)16 * 1048576); // 8 MB [b,h,d,s]
  _Float16* Of = (_Float16*)(ws + (size_t)24 * 1048576); // 8 MB [b,s,D]

  // QKV: M=4096, N=3072, 128x128 tiles
  qkv_gemm_kernel<<<dim3(3072 / 128, 4096 / 128), 256, 0, stream>>>(
      X, Wqkv, bqkv, Qh, Kh, Vt);

  // Flash attention: 4096 wave-tiles / 8 waves per block
  flash_attn_kernel<<<dim3(512), 256, 0, stream>>>(Qh, Kh, Vt, Of);

  // Projection: M=4096, N=1024, 128x128 tiles
  proj_gemm_kernel<<<dim3(1024 / 128, 4096 / 128), 256, 0, stream>>>(
      Of, Wp, bp, out);
}
